// DS_Block_1915555414515
// MI455X (gfx1250) — matrix operand order repaired and checked by reading, not compiled, not run on hardware
//
#include <hip/hip_runtime.h>
#include <cstddef>
#include <cstdint>

// ---------------------------------------------------------------------------
// DS block for MI455X (gfx1250, wave32).
// GEMMs (l1, l2: 128x128 @ 128x4096 per batch) run on v_wmma_f32_16x16x32_f16.
// Activations are stored point-major [n][c] (f16) so both A and B WMMA
// fragments are contiguous 128-bit LDS loads.  B panels are staged
// global->LDS with the Tensor Data Mover (tensor_load_to_lds +
// s_wait_tensorcnt), A is staged with b128 loads.
// ---------------------------------------------------------------------------

#define B_   32
#define N_   8192
#define NDS_ 4096
#define C_   128

typedef _Float16 v16h __attribute__((ext_vector_type(16)));
typedef _Float16 v8h  __attribute__((ext_vector_type(8)));
typedef float    v8f  __attribute__((ext_vector_type(8)));
typedef uint32_t u32x4 __attribute__((ext_vector_type(4)));
typedef uint32_t u32x8 __attribute__((ext_vector_type(8)));

// ---- output layout (flat float offsets, reference return order) ----
constexpr size_t O_XDS  = 0;                                   // (32,1,4096,4)
constexpr size_t O_YDS  = O_XDS  + (size_t)B_ * NDS_ * 4;      // (32,4096)
constexpr size_t O_W0   = O_YDS  + (size_t)B_ * NDS_;          // (32,8192)
constexpr size_t O_W1   = O_W0   + (size_t)B_ * N_;            // (32,8192)
constexpr size_t O_W2M  = O_W1   + (size_t)B_ * N_;            // (32,4096)
constexpr size_t O_W0DS = O_W2M  + (size_t)B_ * NDS_;          // (32,4096)
constexpr size_t O_W1DS = O_W0DS + (size_t)B_ * NDS_;          // (32,4096)
constexpr size_t O_EHAT = O_W1DS + (size_t)B_ * NDS_;          // (32,9)

// ---------------------------------------------------------------------------
// TDM: 2D tile load global -> LDS (one instruction per wave).
// D# group0/group1 packed per CDNA5 ISA.
// Units are elements of 2 bytes (data_size=1).
// ---------------------------------------------------------------------------
__device__ __forceinline__ void tdm_load_2d_f16(uint32_t lds_off_bytes,
                                                const void* gaddr,
                                                uint32_t tile_w, uint32_t tile_h,
                                                uint32_t tensor_w, uint32_t tensor_h,
                                                uint32_t row_stride) {
  const uint64_t ga = (uint64_t)(uintptr_t)gaddr;
  u32x4 g0;
  g0[0] = 1u;                                              // count=1, user mode
  g0[1] = lds_off_bytes;                                   // lds_addr
  g0[2] = (uint32_t)ga;                                    // global_addr[31:0]
  g0[3] = (uint32_t)((ga >> 32) & 0x01FFFFFFu)             // global_addr[56:32]
          | 0x80000000u;                                   // type=2 ("image")
  u32x8 g1;
  g1[0] = 0x00010000u;                                     // data_size=1 (2B)
  g1[1] = (tensor_w & 0xFFFFu) << 16;                      // tensor_dim0[15:0]
  g1[2] = ((tensor_w >> 16) & 0xFFFFu)                     // tensor_dim0[31:16]
          | ((tensor_h & 0xFFFFu) << 16);                  // tensor_dim1[15:0]
  g1[3] = ((tensor_h >> 16) & 0xFFFFu)                     // tensor_dim1[31:16]
          | ((tile_w & 0xFFFFu) << 16);                    // tile_dim0
  g1[4] = (tile_h & 0xFFFFu);                              // tile_dim1 (tile_dim2=0)
  g1[5] = row_stride;                                      // tensor_dim0_stride[31:0]
  g1[6] = 0u;                                              // stride hi / dim1 stride
  g1[7] = 0u;
  asm volatile("tensor_load_to_lds %0, %1" :: "s"(g0), "s"(g1) : "memory");
}

// ===========================================================================
// 1) per-point conv+bn0+relu, then the two 1x128 scoring heads -> w0, w1
// ===========================================================================
__global__ void k_conv_scores(const float* __restrict__ x,
                              const float* __restrict__ conv_w,
                              const float* __restrict__ conv_b,
                              const float* __restrict__ bn0g,
                              const float* __restrict__ bn0b,
                              const float* __restrict__ lin0w,
                              const float* __restrict__ lin0b,
                              const float* __restrict__ lin1w,
                              const float* __restrict__ lin1b,
                              float* __restrict__ out) {
  size_t i = (size_t)blockIdx.x * blockDim.x + threadIdx.x;   // b*N + n
  if (i >= (size_t)B_ * N_) return;
  const float* xp = x + i * 4;
  float a0 = xp[0], a1 = xp[1], a2 = xp[2], a3 = xp[3];
  float s0 = 0.f, s1 = 0.f;
  #pragma unroll 4
  for (int c = 0; c < C_; ++c) {
    float f = conv_w[c * 4 + 0] * a0 + conv_w[c * 4 + 1] * a1 +
              conv_w[c * 4 + 2] * a2 + conv_w[c * 4 + 3] * a3 + conv_b[c];
    f = fmaxf(f * bn0g[c] + bn0b[c], 0.f);
    s0 += lin0w[c] * f;
    s1 += lin1w[c] * f;
  }
  out[O_W0 + i] = s0 + lin0b[0];
  out[O_W1 + i] = s1 + lin1b[0];
}

// ===========================================================================
// 2) weight f32 -> f16 conversion for WMMA A-operands
// ===========================================================================
__global__ void k_cvt_weights(const float* __restrict__ l1w,
                              const float* __restrict__ l2w,
                              _Float16* __restrict__ w1h,
                              _Float16* __restrict__ w2h) {
  int i = blockIdx.x * blockDim.x + threadIdx.x;
  if (i < C_ * C_) {
    w1h[i] = (_Float16)l1w[i];
    w2h[i] = (_Float16)l2w[i];
  }
}

// ===========================================================================
// 3) per-batch bitonic argsort (descending on w1) in LDS, top-4096 gather,
//    and recompute feat_ds (f16, point-major [n][c]) for gathered points.
// ===========================================================================
__global__ void __launch_bounds__(1024)
k_sort_gather(const float* __restrict__ x,
              const float* __restrict__ y,
              const float* __restrict__ conv_w,
              const float* __restrict__ conv_b,
              const float* __restrict__ bn0g,
              const float* __restrict__ bn0b,
              float* __restrict__ out,
              int* __restrict__ idx_ws,
              _Float16* __restrict__ feath) {
  __shared__ float skey[N_];
  __shared__ int   sidx[N_];
  const int b   = blockIdx.x;
  const int tid = threadIdx.x;
  for (int i = tid; i < N_; i += 1024) {
    skey[i] = out[O_W1 + (size_t)b * N_ + i];
    sidx[i] = i;
  }
  __syncthreads();
  // bitonic sort, final direction descending
  for (int k = 2; k <= N_; k <<= 1) {
    for (int j = k >> 1; j > 0; j >>= 1) {
      for (int t = tid; t < N_; t += 1024) {
        int l = t ^ j;
        if (l > t) {
          bool desc = ((t & k) == 0);
          float a = skey[t], c = skey[l];
          bool sw = desc ? (a < c) : (a > c);
          if (sw) {
            skey[t] = c; skey[l] = a;
            int tm = sidx[t]; sidx[t] = sidx[l]; sidx[l] = tm;
          }
        }
      }
      __syncthreads();
    }
  }
  // gather top NDS_ and recompute feat (point-major) for those points
  for (int i = tid; i < NDS_; i += 1024) {
    int p = sidx[i];
    size_t di = (size_t)b * NDS_ + i;
    idx_ws[di] = p;
    out[O_W1DS + di] = skey[i];
    out[O_W0DS + di] = out[O_W0 + (size_t)b * N_ + p];
    out[O_YDS  + di] = y[(size_t)b * N_ + p];
    const float* xp = x + ((size_t)b * N_ + p) * 4;
    float a0 = xp[0], a1 = xp[1], a2 = xp[2], a3 = xp[3];
    float* xo = out + O_XDS + di * 4;
    xo[0] = a0; xo[1] = a1; xo[2] = a2; xo[3] = a3;
    _Float16* fp = feath + di * C_;
    #pragma unroll 2
    for (int c0 = 0; c0 < C_; c0 += 8) {
      v8h pk;
      #pragma unroll
      for (int u = 0; u < 8; ++u) {
        int c = c0 + u;
        float f = conv_w[c * 4 + 0] * a0 + conv_w[c * 4 + 1] * a1 +
                  conv_w[c * 4 + 2] * a2 + conv_w[c * 4 + 3] * a3 + conv_b[c];
        pk[u] = (_Float16)fmaxf(f * bn0g[c] + bn0b[c], 0.f);
      }
      *(v8h*)(fp + c0) = pk;
    }
  }
}

// ===========================================================================
// 4) WMMA GEMM: Out[b, 0:128, n0:n0+32] = A(128x128) * act[b](4096 x 128)^T
//    block = 8 waves (one m-tile each).  A staged via b128 loads, B panel
//    (32 points x 128 ch, point-major) staged via TDM (4 points per wave).
//    Both A and B fragments are contiguous 128-bit LDS loads.
// ===========================================================================
__global__ void __launch_bounds__(256)
k_gemm(const _Float16* __restrict__ A,      // 128x128 row-major (M x K)
       const _Float16* __restrict__ Bm,     // per-batch [n][c] = 4096 x 128
       const float* __restrict__ bias,
       float* __restrict__ Out) {            // per-batch [c][n] f32
  __shared__ _Float16 ldsA[C_ * C_];        // 32 KB, [m][k]
  __shared__ _Float16 ldsB[32 * C_];        // 8 KB,  [n_local][c]
  const int tid  = threadIdx.x;
  const int wv   = tid >> 5;                // wave id = m-tile
  const int lane = tid & 31;
  const int hi   = (lane & 16) ? 1 : 0;
  const int lm   = lane & 15;
  const int n0   = blockIdx.x * 32;
  const int b    = blockIdx.z;
  const _Float16* Bb = Bm + (size_t)b * NDS_ * C_;

  // stage A (whole 128x128) with coalesced 16-byte copies
  {
    const uint4* Ag = (const uint4*)A;
    uint4* Al = (uint4*)ldsA;
    #pragma unroll
    for (int i = 0; i < 8; ++i) Al[tid + i * 256] = Ag[tid + i * 256];
  }
  // stage B panel: wave wv DMAs points [n0+wv*4, n0+wv*4+4), 128 ch each
  tdm_load_2d_f16((uint32_t)(uintptr_t)(ldsB + wv * 4 * C_),
                  Bb + (size_t)(n0 + wv * 4) * C_,
                  /*tile_w=*/C_, /*tile_h=*/4,
                  /*tensor_w=*/C_, /*tensor_h=*/NDS_,
                  /*row_stride=*/C_);
  __builtin_amdgcn_s_wait_tensorcnt(0);
  __syncthreads();

  v8f acc0 = {}, acc1 = {};
  const int m0 = wv * 16;
  #pragma unroll
  for (int ks = 0; ks < 4; ++ks) {
    const int kb = ks * 32;
    // A fragment 16x32: two contiguous 8-half chunks per lane
    v16h av, bv0, bv1;
    const _Float16* Ar = ldsA + (size_t)(m0 + lm) * C_ + kb + hi * 8;
    #pragma unroll
    for (int i = 0; i < 8; ++i) {
      int k = (i & 3) * 2 + (i >> 2) * 16;
      av[2 * i]     = Ar[k];
      av[2 * i + 1] = Ar[k + 1];
    }
    // B fragments 32x16: lane reads two 8-half chunks along K (table order)
    const _Float16* Bc0 = ldsB + (size_t)lm * C_ + kb + hi * 8;
    const _Float16* Bc1 = Bc0 + 16 * C_;
    #pragma unroll
    for (int j = 0; j < 16; ++j) {
      bv0[j] = Bc0[(j & 7) + ((j >> 3) << 4)];
      bv1[j] = Bc1[(j & 7) + ((j >> 3) << 4)];
    }
    acc0 = __builtin_amdgcn_wmma_f32_16x16x32_f16(false, av, false, bv0,
                                                  (short)0, acc0, false, false);
    acc1 = __builtin_amdgcn_wmma_f32_16x16x32_f16(false, av, false, bv1,
                                                  (short)0, acc1, false, false);
  }
  #pragma unroll
  for (int r = 0; r < 8; ++r) {
    int mm = m0 + r + hi * 8;
    float bz = bias[mm];
    float* op = Out + ((size_t)b * C_ + mm) * NDS_ + n0 + lm;
    op[0]  = acc0[r] + bz;
    op[16] = acc1[r] + bz;
  }
}

// ===========================================================================
// 5) per-(b,c) mean/var over 4096 points (instance norm stats)
// ===========================================================================
__global__ void k_meanvar(const float* __restrict__ H,
                          float* __restrict__ mu, float* __restrict__ var) {
  const int c = blockIdx.x, b = blockIdx.y;
  const float* row = H + ((size_t)b * C_ + c) * NDS_;
  float s = 0.f, s2 = 0.f;
  for (int i = threadIdx.x; i < NDS_; i += 256) {
    float v = row[i]; s += v; s2 += v * v;
  }
  __shared__ float ss[256], qq[256];
  ss[threadIdx.x] = s; qq[threadIdx.x] = s2;
  __syncthreads();
  for (int st = 128; st > 0; st >>= 1) {
    if (threadIdx.x < st) {
      ss[threadIdx.x] += ss[threadIdx.x + st];
      qq[threadIdx.x] += qq[threadIdx.x + st];
    }
    __syncthreads();
  }
  if (threadIdx.x == 0) {
    float m = ss[0] * (1.f / NDS_);
    mu[b * C_ + c]  = m;
    var[b * C_ + c] = qq[0] * (1.f / NDS_) - m * m;
  }
}

// ===========================================================================
// 6) instnorm + bn1 + relu -> f16 point-major.  One thread = 8 channels of
//    one point: f32 reads stay lane-coalesced, f16 write is one 16B store.
// ===========================================================================
__global__ void k_norm1(const float* __restrict__ H,
                        const float* __restrict__ mu, const float* __restrict__ var,
                        const float* __restrict__ g, const float* __restrict__ bb,
                        _Float16* __restrict__ Oh) {
  size_t i = (size_t)blockIdx.x * 256 + threadIdx.x;   // (b, group, n)
  int n  = (int)(i & (NDS_ - 1));
  int gr = (int)((i >> 12) & 15);
  int b  = (int)(i >> 16);
  int c0 = gr * 8;
  v8h pk;
  #pragma unroll
  for (int u = 0; u < 8; ++u) {
    int c = c0 + u;
    size_t bc = (size_t)b * C_ + c;
    float hn = (H[bc * NDS_ + n] - mu[bc]) * rsqrtf(var[bc] + 1e-5f);
    pk[u] = (_Float16)fmaxf(hn * g[c] + bb[c], 0.f);
  }
  *(v8h*)(Oh + ((size_t)b * NDS_ + n) * C_ + c0) = pk;
}

// ===========================================================================
// 7) instnorm + bn2, residual add feat_ds (point-major f16), relu ->
//    feat2 in-place into H (channel-major f32).
// ===========================================================================
__global__ void k_norm2(float* __restrict__ H,
                        const float* __restrict__ mu, const float* __restrict__ var,
                        const float* __restrict__ g, const float* __restrict__ bb,
                        const _Float16* __restrict__ feath) {
  size_t i = (size_t)blockIdx.x * 256 + threadIdx.x;   // (b, group, n)
  int n  = (int)(i & (NDS_ - 1));
  int gr = (int)((i >> 12) & 15);
  int b  = (int)(i >> 16);
  int c0 = gr * 8;
  v8h fk = *(const v8h*)(feath + ((size_t)b * NDS_ + n) * C_ + c0);
  #pragma unroll
  for (int u = 0; u < 8; ++u) {
    int c = c0 + u;
    size_t bc = (size_t)b * C_ + c;
    size_t hidx = bc * NDS_ + n;
    float hn = (H[hidx] - mu[bc]) * rsqrtf(var[bc] + 1e-5f);
    float h2 = hn * g[c] + bb[c];
    H[hidx] = fmaxf(h2 + (float)fk[u], 0.f);
  }
}

// ===========================================================================
// 8) lin2 head (2x128 dot per point), mask/weight
// ===========================================================================
__global__ void k_w2(const float* __restrict__ F2,
                     const float* __restrict__ lin2w, const float* __restrict__ lin2b,
                     float* __restrict__ out, float* __restrict__ wbuf) {
  size_t i = (size_t)blockIdx.x * 256 + threadIdx.x;   // b*NDS + n
  int b = (int)(i >> 12);
  int n = (int)(i & (NDS_ - 1));
  float s0 = lin2b[0], s1 = lin2b[1];
  const float* Fb = F2 + (size_t)b * C_ * NDS_ + n;
  #pragma unroll 4
  for (int c = 0; c < C_; ++c) {
    float f = Fb[(size_t)c * NDS_];
    s0 += lin2w[c] * f;
    s1 += lin2w[C_ + c] * f;
  }
  out[O_W2M + i] = s0;
  float mask = 1.f / (1.f + expf(-s0));
  wbuf[i] = expf(s1) * mask;
}

// ===========================================================================
// 9) per-batch weight sum (and zero XwX accumulators)
// ===========================================================================
__global__ void k_wsum(const float* __restrict__ wbuf,
                       float* __restrict__ wsum, float* __restrict__ xwx) {
  const int b = blockIdx.x;
  float s = 0.f;
  for (int i = threadIdx.x; i < NDS_; i += 256) s += wbuf[(size_t)b * NDS_ + i];
  __shared__ float ss[256];
  ss[threadIdx.x] = s;
  __syncthreads();
  for (int st = 128; st > 0; st >>= 1) {
    if (threadIdx.x < st) ss[threadIdx.x] += ss[threadIdx.x + st];
    __syncthreads();
  }
  if (threadIdx.x == 0) wsum[b] = ss[0];
  if (threadIdx.x < 81) xwx[b * 81 + threadIdx.x] = 0.f;
}

// ===========================================================================
// 10) XwX = sum_n w_n X_n X_n^T  (9x9, upper triangle, float atomics)
// ===========================================================================
__global__ void k_xwx(const float* __restrict__ xds,   // out + O_XDS
                      const float* __restrict__ wbuf,
                      const float* __restrict__ wsum,
                      float* __restrict__ xwx) {
  const int b = blockIdx.x;
  float p[45];
  #pragma unroll
  for (int t = 0; t < 45; ++t) p[t] = 0.f;
  const float inv = 1.f / (wsum[b] + 1e-5f);
  for (int n = threadIdx.x; n < NDS_; n += 256) {
    const float* xp = xds + ((size_t)b * NDS_ + n) * 4;
    float a0 = xp[0], a1 = xp[1], a2 = xp[2], a3 = xp[3];
    float X[9] = {a2 * a0, a2 * a1, a2, a3 * a0, a3 * a1, a3, a0, a1, 1.f};
    float w = wbuf[(size_t)b * NDS_ + n] * inv;
    int t = 0;
    #pragma unroll
    for (int ii = 0; ii < 9; ++ii) {
      float wx = w * X[ii];
      #pragma unroll
      for (int jj = ii; jj < 9; ++jj) p[t++] += wx * X[jj];
    }
  }
  int t = 0;
  #pragma unroll
  for (int ii = 0; ii < 9; ++ii)
    #pragma unroll
    for (int jj = ii; jj < 9; ++jj)
      atomicAdd(&xwx[b * 81 + ii * 9 + jj], p[t++]);
}

// ===========================================================================
// 11) 9x9 symmetric Jacobi eigensolver, smallest-eigenvalue eigenvector
// ===========================================================================
__global__ void k_eigh(const float* __restrict__ xwx, float* __restrict__ out) {
  if (threadIdx.x != 0) return;
  const int b = blockIdx.x;
  float M[9][9], V[9][9];
  for (int i = 0; i < 9; ++i)
    for (int j = 0; j < 9; ++j) {
      float v = (j >= i) ? xwx[b * 81 + i * 9 + j] : xwx[b * 81 + j * 9 + i];
      M[i][j] = v;
      V[i][j] = (i == j) ? 1.f : 0.f;
    }
  for (int sweep = 0; sweep < 24; ++sweep) {
    for (int pi = 0; pi < 8; ++pi) {
      for (int q = pi + 1; q < 9; ++q) {
        float apq = M[pi][q];
        if (fabsf(apq) < 1e-18f) continue;
        float theta = 0.5f * (M[q][q] - M[pi][pi]) / apq;
        float tt = ((theta >= 0.f) ? 1.f : -1.f) /
                   (fabsf(theta) + sqrtf(theta * theta + 1.f));
        float c = 1.f / sqrtf(tt * tt + 1.f);
        float s = tt * c;
        for (int k = 0; k < 9; ++k) {
          float mkp = M[k][pi], mkq = M[k][q];
          M[k][pi] = c * mkp - s * mkq;
          M[k][q]  = s * mkp + c * mkq;
        }
        for (int k = 0; k < 9; ++k) {
          float mpk = M[pi][k], mqk = M[q][k];
          M[pi][k] = c * mpk - s * mqk;
          M[q][k]  = s * mpk + c * mqk;
        }
        for (int k = 0; k < 9; ++k) {
          float vkp = V[k][pi], vkq = V[k][q];
          V[k][pi] = c * vkp - s * vkq;
          V[k][q]  = s * vkp + c * vkq;
        }
      }
    }
  }
  int jm = 0;
  float dm = M[0][0];
  for (int j = 1; j < 9; ++j)
    if (M[j][j] < dm) { dm = M[j][j]; jm = j; }
  float nn = 0.f;
  for (int k = 0; k < 9; ++k) nn += V[k][jm] * V[k][jm];
  float inv = rsqrtf(fmaxf(nn, 1e-30f));
  for (int k = 0; k < 9; ++k) out[O_EHAT + b * 9 + k] = V[k][jm] * inv;
}

// ===========================================================================
extern "C" void kernel_launch(void* const* d_in, const int* in_sizes, int n_in,
                              void* d_out, int out_size, void* d_ws, size_t ws_size,
                              hipStream_t stream) {
  const float* x      = (const float*)d_in[0];
  const float* y      = (const float*)d_in[1];
  const float* conv_w = (const float*)d_in[2];
  const float* conv_b = (const float*)d_in[3];
  const float* bn0_g  = (const float*)d_in[4];
  const float* bn0_b  = (const float*)d_in[5];
  const float* lin0_w = (const float*)d_in[6];
  const float* lin0_b = (const float*)d_in[7];
  const float* lin1_w = (const float*)d_in[8];
  const float* lin1_b = (const float*)d_in[9];
  const float* l1_w   = (const float*)d_in[10];
  const float* l1_b   = (const float*)d_in[11];
  const float* bn1_g  = (const float*)d_in[12];
  const float* bn1_b  = (const float*)d_in[13];
  const float* l2_w   = (const float*)d_in[14];
  const float* l2_b   = (const float*)d_in[15];
  const float* bn2_g  = (const float*)d_in[16];
  const float* bn2_b  = (const float*)d_in[17];
  const float* lin2_w = (const float*)d_in[18];
  const float* lin2_b = (const float*)d_in[19];
  float* out = (float*)d_out;

  // workspace carve-up
  auto al = [](size_t v) { return (v + 255) & ~(size_t)255; };
  uint8_t* ws = (uint8_t*)d_ws;
  size_t off = 0;
  int* idx_ws = (int*)(ws + off);           off = al(off + (size_t)B_ * NDS_ * 4);
  float* wbuf = (float*)(ws + off);         off = al(off + (size_t)B_ * NDS_ * 4);
  float* wsum = (float*)(ws + off);         off = al(off + (size_t)B_ * 4);
  float* xwx  = (float*)(ws + off);         off = al(off + (size_t)B_ * 81 * 4);
  float* mu   = (float*)(ws + off);         off = al(off + (size_t)B_ * C_ * 4);
  float* var  = (float*)(ws + off);         off = al(off + (size_t)B_ * C_ * 4);
  _Float16* w1h   = (_Float16*)(ws + off);  off = al(off + (size_t)C_ * C_ * 2);
  _Float16* w2h   = (_Float16*)(ws + off);  off = al(off + (size_t)C_ * C_ * 2);
  _Float16* feath = (_Float16*)(ws + off);  off = al(off + (size_t)B_ * NDS_ * C_ * 2);
  _Float16* h1h   = (_Float16*)(ws + off);  off = al(off + (size_t)B_ * NDS_ * C_ * 2);
  float* hpre     = (float*)(ws + off);     off = al(off + (size_t)B_ * C_ * NDS_ * 4);

  // 1) conv + heads -> w0, w1
  k_conv_scores<<<(B_ * N_) / 256, 256, 0, stream>>>(
      x, conv_w, conv_b, bn0_g, bn0_b, lin0_w, lin0_b, lin1_w, lin1_b, out);
  // 2) weights to f16
  k_cvt_weights<<<(C_ * C_ + 255) / 256, 256, 0, stream>>>(l1_w, l2_w, w1h, w2h);
  // 3) per-batch sort + gather + feat_ds recompute (point-major)
  k_sort_gather<<<B_, 1024, 0, stream>>>(x, y, conv_w, conv_b, bn0_g, bn0_b,
                                         out, idx_ws, feath);
  // 4) layer 1 GEMM (WMMA + TDM)
  k_gemm<<<dim3(NDS_ / 32, 1, B_), 256, 0, stream>>>(w1h, feath, l1_b, hpre);
  // 5) instnorm stats
  k_meanvar<<<dim3(C_, B_), 256, 0, stream>>>(hpre, mu, var);
  // 6) norm1 + bn1 + relu -> f16 point-major
  k_norm1<<<(size_t)B_ * 16 * NDS_ / 256, 256, 0, stream>>>(hpre, mu, var,
                                                            bn1_g, bn1_b, h1h);
  // 7) layer 2 GEMM (WMMA + TDM)
  k_gemm<<<dim3(NDS_ / 32, 1, B_), 256, 0, stream>>>(w2h, h1h, l2_b, hpre);
  // 8) instnorm stats (reuse mu/var)
  k_meanvar<<<dim3(C_, B_), 256, 0, stream>>>(hpre, mu, var);
  // 9) norm2 + bn2 + residual + relu -> feat2 (in place)
  k_norm2<<<(size_t)B_ * 16 * NDS_ / 256, 256, 0, stream>>>(hpre, mu, var,
                                                            bn2_g, bn2_b, feath);
  // 10) lin2 head, mask, weights
  k_w2<<<(B_ * NDS_) / 256, 256, 0, stream>>>(hpre, lin2_w, lin2_b, out, wbuf);
  // 11) weight sums + zero XwX
  k_wsum<<<B_, 256, 0, stream>>>(wbuf, wsum, xwx);
  // 12) weighted moment matrix
  k_xwx<<<B_, 256, 0, stream>>>(out + O_XDS, wbuf, wsum, xwx);
  // 13) 9x9 eigensolve -> e_hat
  k_eigh<<<B_, 32, 0, stream>>>(xwx, out);
  (void)in_sizes; (void)n_in; (void)out_size; (void)ws_size;
}
